// RWKV_Tmix_x070_83743272338045
// MI455X (gfx1250) — matrix operand order repaired and checked by reading, not compiled, not run on hardware
//
#include <hip/hip_runtime.h>
#include <hip/hip_bf16.h>
#include <cstdint>
#include <cstddef>

// ---------------- problem constants ----------------
#define B_ 8
#define T_ 1024
#define C_ 1024
#define H_ 16
#define HS_ 64
#define DW_ 64
#define DA_ 64
#define DV_ 32
#define DG_ 160
#define EPS_ 0.00064f

typedef __bf16 bf16;
typedef __attribute__((ext_vector_type(16))) __bf16 v16bf;
typedef __attribute__((ext_vector_type(8)))  __bf16 v8bf;
typedef __attribute__((ext_vector_type(8)))  float  v8f;
typedef __attribute__((ext_vector_type(4)))  unsigned v4u;
typedef __attribute__((ext_vector_type(8)))  unsigned v8u;

// Low 32 bits of a generic pointer to __shared__ == AS3 (LDS) byte address
// (flat LDS aperture truncates to addr[31:0]; HW adds the wave's LDS_BASE).
__device__ __forceinline__ unsigned lds_addr32(const void* p) {
    return (unsigned)(uintptr_t)p;
}

// CDNA5 async global->LDS DMA (ASYNCcnt-tracked), per-lane 16B / 4B.
__device__ __forceinline__ void async_load_b128(unsigned lds_dst, const void* gsrc) {
    asm volatile("global_load_async_to_lds_b128 %0, %1, off"
                 :: "v"(lds_dst), "v"(gsrc) : "memory");
}
__device__ __forceinline__ void async_load_b32(unsigned lds_dst, const void* gsrc) {
    asm volatile("global_load_async_to_lds_b32 %0, %1, off"
                 :: "v"(lds_dst), "v"(gsrc) : "memory");
}
__device__ __forceinline__ void wait_async0() {
    asm volatile("s_wait_asynccnt 0x0" ::: "memory");
}

// =====================================================================
// Weight prep: fp32 -> bf16, with optional transpose into [N,K] layout
// =====================================================================
__global__ __launch_bounds__(256) void convert_bf16_kernel(
    bf16* __restrict__ dst, const float* __restrict__ src, size_t n)
{
    size_t g = (size_t)blockIdx.x * blockDim.x + threadIdx.x;
    if (g < n) dst[g] = (bf16)src[g];
}

// src laid out [K, N] row-major; dst[n*K + k] = src[k*N + n]
__global__ __launch_bounds__(256) void transpose_to_nk_kernel(
    bf16* __restrict__ dst, const float* __restrict__ src, int K, int N)
{
    size_t g = (size_t)blockIdx.x * blockDim.x + threadIdx.x;
    size_t total = (size_t)K * N;
    if (g >= total) return;
    int k = (int)(g % K);
    int n = (int)(g / K);
    dst[g] = (bf16)src[(size_t)k * N + n];
}

// =====================================================================
// Token shift: xx = x[t-1]-x[t]; emit six bf16 mixed activations
// =====================================================================
__global__ __launch_bounds__(256) void token_shift_kernel(
    const float* __restrict__ x,
    const float* __restrict__ cr, const float* __restrict__ cw,
    const float* __restrict__ ck, const float* __restrict__ cv,
    const float* __restrict__ ca, const float* __restrict__ cg,
    bf16* __restrict__ oxr, bf16* __restrict__ oxw, bf16* __restrict__ oxk,
    bf16* __restrict__ oxv, bf16* __restrict__ oxa, bf16* __restrict__ oxg)
{
    size_t gid = (size_t)blockIdx.x * blockDim.x + threadIdx.x;
    if (gid >= (size_t)B_ * T_ * C_) return;
    int c = (int)(gid & (C_ - 1));
    size_t bt = gid >> 10;              // /C_
    int t = (int)(bt & (T_ - 1));
    float xv = x[gid];
    float xp = (t > 0) ? x[gid - C_] : 0.f;
    float d = xp - xv;
    oxr[gid] = (bf16)(xv + d * cr[c]);
    oxw[gid] = (bf16)(xv + d * cw[c]);
    oxk[gid] = (bf16)(xv + d * ck[c]);
    oxv[gid] = (bf16)(xv + d * cv[c]);
    oxa[gid] = (bf16)(xv + d * ca[c]);
    oxg[gid] = (bf16)(xv + d * cg[c]);
}

// =====================================================================
// Generic WMMA GEMM:  C[M,N] = act( A[M,K] @ Bw[N,K]^T )
//   A, Bw bf16 row-major.  256 threads, 64x64 tile, K-step 32 (== WMMA K).
//   A tile: Tensor Data Mover (tensor_load_to_lds), double-buffered on
//   TENSORcnt, with D# LDS padding (16 DW data + 4 DW pad per row) matching
//   the As[..][40] fragment layout.  B tile: sync load + transpose.
//   act: 0=none 1=tanh 2=sigmoid.  out_bf16: epilogue writes bf16.
// =====================================================================
__global__ __launch_bounds__(256) void gemm_wmma_bf16_kernel(
    const bf16* __restrict__ A, const bf16* __restrict__ Bw,
    void* __restrict__ Cd, int M, int N, int K, int act, int out_bf16)
{
    __shared__ bf16 As[2][64][40];  // double-buffered TDM dest (row = 32+8 pad)
    __shared__ bf16 Bs[32][72];     // 32 K-rows x 64 N (pad->72)

    const int tid  = threadIdx.x;
    const int lane = tid & 31;
    const int wv   = tid >> 5;
    const int wm   = wv & 3;
    const int wn   = wv >> 2;
    const int bm   = blockIdx.y;
    const int bn   = blockIdx.x;

    v8f acc[2];
    #pragma unroll
    for (int e = 0; e < 8; ++e) { acc[0][e] = 0.f; acc[1][e] = 0.f; }

    const int lr = tid >> 2;          // 0..63
    const int lc = (tid & 3) * 8;     // 0,8,16,24

    // ---- TDM: issue one 64x32 bf16 tile DMA (wave-level; EXEC ignored).
    // D# group0: count=1 | lds_addr | global_addr[56:0] | type=2
    // D# group1: data_size=2B, pad 4 DW per 16 DW, dims/strides of [M,K]
    auto tdm_issue_A = [&](int k0, unsigned lds_base) {
        uint64_t gaddr = (uint64_t)(uintptr_t)(A + (size_t)bm * 64 * K + k0);
        v4u g0, gz;
        v8u g1;
        g0[0] = 1u;                                   // count=1, user mode
        g0[1] = lds_base;                             // LDS byte address
        g0[2] = (unsigned)gaddr;                      // global_addr[31:0]
        g0[3] = (unsigned)((gaddr >> 32) & 0x01ffffffULL) | (2u << 30); // +type=2
        g1[0] = (1u << 16)                            // data_size = 2 bytes
              | (1u << 20)                            // pad_enable
              | (3u << 22)                            // pad_interval: 16 DWORDs
              | (3u << 25);                           // pad_amount:   4 DWORDs
        g1[1] = ((unsigned)K & 0xffffu) << 16;        // tensor_dim0[15:0]
        g1[2] = (((unsigned)K >> 16) & 0xffffu)       // tensor_dim0[31:16]
              | (((unsigned)M & 0xffffu) << 16);      // tensor_dim1[15:0]
        g1[3] = (((unsigned)M >> 16) & 0xffffu)       // tensor_dim1[31:16]
              | (32u << 16);                          // tile_dim0 = 32
        g1[4] = 64u;                                  // tile_dim1 = 64, tile_dim2=0
        g1[5] = (unsigned)K;                          // tensor_dim0_stride[31:0]
        g1[6] = 0u;                                   // stride0 hi, stride1 lo
        g1[7] = 0u;                                   // stride1 hi
        #pragma unroll
        for (int e = 0; e < 4; ++e) gz[e] = 0u;       // unused groups 2/3
        asm volatile("tensor_load_to_lds %0, %1, %2, %3"
                     :: "s"(g0), "s"(g1), "s"(gz), "s"(gz) : "memory");
    };

    if (wv == 0) tdm_issue_A(0, lds_addr32(&As[0][0][0]));

    const int nk = K >> 5;
    for (int ks = 0; ks < nk; ++ks) {
        const int k0  = ks << 5;
        const int cur = ks & 1;

        // ---- B tile: sync load + transpose [N,K] -> Bs[K][N]
        {
            const int gn = bn * 64 + lr;
            v8bf bvv;
            if (gn < N) {
                bvv = *reinterpret_cast<const v8bf*>(Bw + (size_t)gn * K + k0 + lc);
                if (k0 + 32 < K)
                    __builtin_prefetch(Bw + (size_t)gn * K + k0 + 32 + lc, 0, 1);
            } else {
                #pragma unroll
                for (int e = 0; e < 8; ++e) bvv[e] = (bf16)0.0f;
            }
            #pragma unroll
            for (int e = 0; e < 8; ++e) Bs[lc + e][lr] = bvv[e];
        }

        if (wv == 0) __builtin_amdgcn_s_wait_tensorcnt(0);  // tile ks landed
        __syncthreads();                                     // publish A+B
        if (wv == 0 && k0 + 32 < K)
            tdm_issue_A(k0 + 32, lds_addr32(&As[cur ^ 1][0][0]));  // prefetch

        // ---- A fragment: lane L (h = L>>4) row = wm*16 + (L&15),
        //      K chunks at cols 8h..8h+7 and 16+8h..23+8h
        const int hh   = lane >> 4;
        const int arow = wm * 16 + (lane & 15);
        const v8bf* ap = reinterpret_cast<const v8bf*>(&As[cur][arow][8 * hh]);
        v8bf alo = ap[0];
        v8bf ahi = ap[2];
        v16bf afrag;
        #pragma unroll
        for (int e = 0; e < 8; ++e) { afrag[e] = alo[e]; afrag[e + 8] = ahi[e]; }

        // ---- B fragments: lane&15 = N column; vector elem = K (8h..8h+7, 16+8h..23+8h)
        #pragma unroll
        for (int sub = 0; sub < 2; ++sub) {
            const int bcol = wn * 32 + sub * 16 + (lane & 15);
            v16bf bfrag;
            #pragma unroll
            for (int e = 0; e < 8; ++e) { bfrag[e] = Bs[8 * hh + e][bcol]; bfrag[e + 8] = Bs[16 + 8 * hh + e][bcol]; }
            acc[sub] = __builtin_amdgcn_wmma_f32_16x16x32_bf16(
                false, afrag, false, bfrag, (short)0, acc[sub], false, false);
        }
        __syncthreads();
    }

    // ---- epilogue: C/D layout: VGPR e -> M = e + 8*(lane>>4), N = lane&15
    const int hh = lane >> 4;
    #pragma unroll
    for (int sub = 0; sub < 2; ++sub) {
        const int col = bn * 64 + wn * 32 + sub * 16 + (lane & 15);
        if (col < N) {
            #pragma unroll
            for (int e = 0; e < 8; ++e) {
                const int row = bm * 64 + wm * 16 + hh * 8 + e;
                float v = acc[sub][e];
                if (act == 1)      v = tanhf(v);
                else if (act == 2) v = 1.f / (1.f + expf(-v));
                if (out_bf16)
                    reinterpret_cast<bf16*>(Cd)[(size_t)row * N + col] = (bf16)v;
                else
                    reinterpret_cast<float*>(Cd)[(size_t)row * N + col] = v;
            }
        }
    }
}

// =====================================================================
// Elementwise mix: per (b,t,h), 64 threads. Produces scan inputs
// *in place* in [B,T,C] layout:
//   wl <- exp(-exp(w));  al <- -kk_hat;  vl <- kk_hat*a;
//   k  <- k*(1+(a-1)*k_a);  v <- v + (v_first-v)*sigmoid(v0+vl_in)
// =====================================================================
__global__ __launch_bounds__(64) void mix_kernel(
    const float* __restrict__ vfirst,
    const float* __restrict__ w0, const float* __restrict__ a0,
    const float* __restrict__ v0,
    const float* __restrict__ k_k, const float* __restrict__ k_a,
    float* __restrict__ k_f, float* __restrict__ v_f,
    float* __restrict__ wl_f, float* __restrict__ al_f, float* __restrict__ vl_f)
{
    const int blk = blockIdx.x;          // (b*T+t)*H + h
    const int h = blk & (H_ - 1);
    const size_t m = (size_t)(blk >> 4);
    const int i = threadIdx.x;
    const int c = h * HS_ + i;
    const size_t idx = m * C_ + c;

    float kraw = k_f[idx];
    float kkv  = kraw * k_k[c];

    __shared__ float red[64];
    red[i] = kkv * kkv;
    __syncthreads();
    #pragma unroll
    for (int s = 32; s > 0; s >>= 1) {
        if (i < s) red[i] += red[i + s];
        __syncthreads();
    }
    float inv = 1.f / fmaxf(sqrtf(red[0]), 1e-12f);
    float kkn = kkv * inv;

    float a_s  = 1.f / (1.f + expf(-(a0[c] + al_f[idx])));
    float wz   = -(w0[c] + wl_f[idx]);
    float wlog = -log1pf(expf(wz)) - 0.5f;          // -softplus(-x) - 0.5
    float dec  = expf(-expf(wlog));
    float vres = 1.f / (1.f + expf(-(v0[c] + vl_f[idx])));
    float vraw = v_f[idx];

    k_f[idx]  = kraw * (1.f + (a_s - 1.f) * k_a[c]);
    v_f[idx]  = vraw + (vfirst[idx] - vraw) * vres;
    wl_f[idx] = dec;
    al_f[idx] = -kkn;
    vl_f[idx] = kkn * a_s;
}

// =====================================================================
// RWKV-7 delta-rule scan. One block per (b,h), 64 threads (2 waves).
// Thread i owns state row i (64 fp32 registers). The six per-step
// vectors are double-buffered in LDS and prefetched one step ahead with
// global_load_async_to_lds_b32 so the async engine overlaps step t+1's
// memory latency with step t's ~320 FMA of VALU work.
// Array slots in each buffer: 0=r 1=w 2=k 3=v 4=a 5=b.
// =====================================================================
__global__ __launch_bounds__(64) void rwkv_scan_kernel(
    const float* __restrict__ r_f, const float* __restrict__ w_f,
    const float* __restrict__ k_f, const float* __restrict__ v_f,
    const float* __restrict__ a_f, const float* __restrict__ b_f,
    float* __restrict__ y_f)
{
    const int bh = blockIdx.x;
    const int b = bh >> 4, h = bh & (H_ - 1);
    const int i = threadIdx.x;

    __shared__ float bufs[2][6 * 64];

    float st[64];
    #pragma unroll
    for (int j = 0; j < 64; ++j) st[j] = 0.f;

    const size_t base = (size_t)b * T_ * C_ + (size_t)h * HS_ + i;

    const unsigned l0 = lds_addr32(&bufs[0][0 * 64 + i]);
    const unsigned l1 = lds_addr32(&bufs[0][1 * 64 + i]);
    const unsigned l2 = lds_addr32(&bufs[0][2 * 64 + i]);
    const unsigned l3 = lds_addr32(&bufs[0][3 * 64 + i]);
    const unsigned l4 = lds_addr32(&bufs[0][4 * 64 + i]);
    const unsigned l5 = lds_addr32(&bufs[0][5 * 64 + i]);
    const unsigned bstride = (unsigned)(6 * 64 * sizeof(float));

    auto issue = [&](int t, int slot) {
        const size_t o = base + (size_t)t * C_;
        const unsigned d = slot ? bstride : 0u;
        async_load_b32(l0 + d, r_f + o);
        async_load_b32(l1 + d, w_f + o);
        async_load_b32(l2 + d, k_f + o);
        async_load_b32(l3 + d, v_f + o);
        async_load_b32(l4 + d, a_f + o);
        async_load_b32(l5 + d, b_f + o);
    };

    issue(0, 0);
    for (int t = 0; t < T_; ++t) {
        wait_async0();           // this wave's batch-t DMA has landed in LDS
        __syncthreads();         // all waves' batches published
        if (t + 1 < T_) issue(t + 1, (t + 1) & 1);

        const float* __restrict__ cb = &bufs[t & 1][0];

        float ad = 0.f;
        #pragma unroll
        for (int j = 0; j < 64; ++j) ad = fmaf(st[j], cb[4 * 64 + j], ad);

        const float vv = cb[3 * 64 + i];
        float y = 0.f;
        #pragma unroll
        for (int j = 0; j < 64; ++j) {
            float s = fmaf(st[j], cb[1 * 64 + j],
                           fmaf(ad, cb[5 * 64 + j], vv * cb[2 * 64 + j]));
            st[j] = s;
            y = fmaf(s, cb[0 * 64 + j], y);
        }
        y_f[base + (size_t)t * C_] = y;
    }
}

// =====================================================================
// Post: per-head GroupNorm + bonus term + gate; emit bf16 for final GEMM
// =====================================================================
__global__ __launch_bounds__(64) void post_kernel(
    const float* __restrict__ y_f, const float* __restrict__ r_f,
    const float* __restrict__ k_f, const float* __restrict__ v_f,
    const float* __restrict__ g_f, const float* __restrict__ r_k,
    const float* __restrict__ ln_g, const float* __restrict__ ln_b,
    bf16* __restrict__ zB)
{
    const int blk = blockIdx.x;
    const int h = blk & (H_ - 1);
    const size_t m = (size_t)(blk >> 4);
    const int i = threadIdx.x;
    const int c = h * HS_ + i;
    const size_t idx = m * C_ + c;

    __shared__ float s1[64], s2[64];
    float yv = y_f[idx];
    s1[i] = yv; s2[i] = yv * yv;
    __syncthreads();
    #pragma unroll
    for (int s = 32; s > 0; s >>= 1) {
        if (i < s) { s1[i] += s1[i + s]; s2[i] += s2[i + s]; }
        __syncthreads();
    }
    float mu  = s1[0] * (1.f / 64.f);
    float var = s2[0] * (1.f / 64.f) - mu * mu;
    __syncthreads();

    float rk = r_f[idx] * k_f[idx] * r_k[c];
    s1[i] = rk;
    __syncthreads();
    #pragma unroll
    for (int s = 32; s > 0; s >>= 1) {
        if (i < s) s1[i] += s1[i + s];
        __syncthreads();
    }
    float dot = s1[0];

    float yn = (yv - mu) * rsqrtf(var + EPS_) * ln_g[c] + ln_b[c];
    float z  = (yn + dot * v_f[idx]) * g_f[idx];
    zB[idx] = (bf16)z;
}

// =====================================================================
// v_first pass-through (second tuple output)
// =====================================================================
__global__ __launch_bounds__(256) void copy_f32_kernel(
    const float* __restrict__ src, float* __restrict__ dst, size_t n)
{
    size_t g = (size_t)blockIdx.x * blockDim.x + threadIdx.x;
    if (g < n) dst[g] = src[g];
}

// =====================================================================
// Host-side orchestration
// =====================================================================
extern "C" void kernel_launch(void* const* d_in, const int* in_sizes, int n_in,
                              void* d_out, int out_size, void* d_ws, size_t ws_size,
                              hipStream_t stream)
{
    (void)in_sizes; (void)n_in; (void)out_size; (void)ws_size;

    const float* x      = (const float*)d_in[0];
    const float* vfirst = (const float*)d_in[1];
    const float* x_r    = (const float*)d_in[2];
    const float* x_w    = (const float*)d_in[3];
    const float* x_k    = (const float*)d_in[4];
    const float* x_v    = (const float*)d_in[5];
    const float* x_a    = (const float*)d_in[6];
    const float* x_g    = (const float*)d_in[7];
    const float* w0     = (const float*)d_in[8];
    const float* w1     = (const float*)d_in[9];
    const float* w2     = (const float*)d_in[10];
    const float* a0     = (const float*)d_in[11];
    const float* a1     = (const float*)d_in[12];
    const float* a2     = (const float*)d_in[13];
    const float* v0     = (const float*)d_in[14];
    const float* v1     = (const float*)d_in[15];
    const float* v2     = (const float*)d_in[16];
    const float* g1     = (const float*)d_in[17];
    const float* g2     = (const float*)d_in[18];
    const float* k_k    = (const float*)d_in[19];
    const float* k_a    = (const float*)d_in[20];
    const float* r_k    = (const float*)d_in[21];
    const float* Wr     = (const float*)d_in[22];
    const float* Wk     = (const float*)d_in[23];
    const float* Wv     = (const float*)d_in[24];
    const float* Wo     = (const float*)d_in[25];
    const float* ln_g   = (const float*)d_in[26];
    const float* ln_b   = (const float*)d_in[27];

    float* outp = (float*)d_out;

    const size_t M  = (size_t)B_ * T_;       // 8192
    const size_t MC = M * C_;                // 8,388,608

    // ---- workspace carve-out
    char* ws = (char*)d_ws;
    size_t off = 0;
    auto take = [&](size_t bytes) -> char* {
        char* p = ws + off;
        off = (off + bytes + 255) & ~(size_t)255;
        return p;
    };

    bf16* xrB = (bf16*)take(MC * 2);
    bf16* xwB = (bf16*)take(MC * 2);
    bf16* xkB = (bf16*)take(MC * 2);
    bf16* xvB = (bf16*)take(MC * 2);
    bf16* xaB = (bf16*)take(MC * 2);
    bf16* xgB = (bf16*)take(MC * 2);

    bf16* WrB = (bf16*)take((size_t)C_ * C_ * 2);
    bf16* WkB = (bf16*)take((size_t)C_ * C_ * 2);
    bf16* WvB = (bf16*)take((size_t)C_ * C_ * 2);
    bf16* WoB = (bf16*)take((size_t)C_ * C_ * 2);

    bf16* w1B = (bf16*)take((size_t)DW_ * C_ * 2);
    bf16* w2B = (bf16*)take((size_t)C_ * DW_ * 2);
    bf16* a1B = (bf16*)take((size_t)DA_ * C_ * 2);
    bf16* a2B = (bf16*)take((size_t)C_ * DA_ * 2);
    bf16* v1B = (bf16*)take((size_t)DV_ * C_ * 2);
    bf16* v2B = (bf16*)take((size_t)C_ * DV_ * 2);
    bf16* g1B = (bf16*)take((size_t)DG_ * C_ * 2);
    bf16* g2B = (bf16*)take((size_t)C_ * DG_ * 2);

    bf16* hwB = (bf16*)take(M * DW_ * 2);
    bf16* haB = (bf16*)take(M * DA_ * 2);
    bf16* hvB = (bf16*)take(M * DV_ * 2);
    bf16* hgB = (bf16*)take(M * DG_ * 2);

    bf16* zB  = (bf16*)take(MC * 2);

    float* r_f  = (float*)take(MC * 4);
    float* k_f  = (float*)take(MC * 4);
    float* v_f  = (float*)take(MC * 4);
    float* wl_f = (float*)take(MC * 4);   // wlin -> decay (in place)
    float* al_f = (float*)take(MC * 4);   // alin -> -kk_hat
    float* vl_f = (float*)take(MC * 4);   // vlin -> kk_hat * a
    float* g_f  = (float*)take(MC * 4);
    float* y_f  = (float*)take(MC * 4);

    // ---- 1. weight prep (bf16, [N,K] layout)
    {
        size_t nCC = (size_t)C_ * C_;
        int g = (int)((nCC + 255) / 256);
        convert_bf16_kernel<<<g, 256, 0, stream>>>(WrB, Wr, nCC);
        convert_bf16_kernel<<<g, 256, 0, stream>>>(WkB, Wk, nCC);
        convert_bf16_kernel<<<g, 256, 0, stream>>>(WvB, Wv, nCC);
        convert_bf16_kernel<<<g, 256, 0, stream>>>(WoB, Wo, nCC);

        auto tgrid = [](size_t n) { return (int)((n + 255) / 256); };
        transpose_to_nk_kernel<<<tgrid((size_t)C_ * DW_), 256, 0, stream>>>(w1B, w1, C_, DW_);
        transpose_to_nk_kernel<<<tgrid((size_t)C_ * DW_), 256, 0, stream>>>(w2B, w2, DW_, C_);
        transpose_to_nk_kernel<<<tgrid((size_t)C_ * DA_), 256, 0, stream>>>(a1B, a1, C_, DA_);
        transpose_to_nk_kernel<<<tgrid((size_t)C_ * DA_), 256, 0, stream>>>(a2B, a2, DA_, C_);
        transpose_to_nk_kernel<<<tgrid((size_t)C_ * DV_), 256, 0, stream>>>(v1B, v1, C_, DV_);
        transpose_to_nk_kernel<<<tgrid((size_t)C_ * DV_), 256, 0, stream>>>(v2B, v2, DV_, C_);
        transpose_to_nk_kernel<<<tgrid((size_t)C_ * DG_), 256, 0, stream>>>(g1B, g1, C_, DG_);
        transpose_to_nk_kernel<<<tgrid((size_t)C_ * DG_), 256, 0, stream>>>(g2B, g2, DG_, C_);
    }

    // ---- 2. token shift -> six bf16 activation tensors
    {
        int g = (int)((MC + 255) / 256);
        token_shift_kernel<<<g, 256, 0, stream>>>(x, x_r, x_w, x_k, x_v, x_a, x_g,
                                                  xrB, xwB, xkB, xvB, xaB, xgB);
    }

    // ---- 3. GEMMs (WMMA bf16, f32 accumulate; A tiles via TDM)
    auto ggrid = [](int N) { return dim3((unsigned)((N + 63) / 64), (unsigned)(8192 / 64)); };

    gemm_wmma_bf16_kernel<<<ggrid(C_),  256, 0, stream>>>(xrB, WrB, r_f, (int)M, C_, C_, 0, 0);
    gemm_wmma_bf16_kernel<<<ggrid(C_),  256, 0, stream>>>(xkB, WkB, k_f, (int)M, C_, C_, 0, 0);
    gemm_wmma_bf16_kernel<<<ggrid(C_),  256, 0, stream>>>(xvB, WvB, v_f, (int)M, C_, C_, 0, 0);

    gemm_wmma_bf16_kernel<<<ggrid(DW_), 256, 0, stream>>>(xwB, w1B, hwB, (int)M, DW_, C_, 1, 1); // tanh
    gemm_wmma_bf16_kernel<<<ggrid(DA_), 256, 0, stream>>>(xaB, a1B, haB, (int)M, DA_, C_, 0, 1);
    gemm_wmma_bf16_kernel<<<ggrid(DV_), 256, 0, stream>>>(xvB, v1B, hvB, (int)M, DV_, C_, 0, 1);
    gemm_wmma_bf16_kernel<<<ggrid(DG_), 256, 0, stream>>>(xgB, g1B, hgB, (int)M, DG_, C_, 2, 1); // sigmoid

    gemm_wmma_bf16_kernel<<<ggrid(C_),  256, 0, stream>>>(hwB, w2B, wl_f, (int)M, C_, DW_, 0, 0);
    gemm_wmma_bf16_kernel<<<ggrid(C_),  256, 0, stream>>>(haB, a2B, al_f, (int)M, C_, DA_, 0, 0);
    gemm_wmma_bf16_kernel<<<ggrid(C_),  256, 0, stream>>>(hvB, v2B, vl_f, (int)M, C_, DV_, 0, 0);
    gemm_wmma_bf16_kernel<<<ggrid(C_),  256, 0, stream>>>(hgB, g2B, g_f,  (int)M, C_, DG_, 0, 0);

    // ---- 4. elementwise mix (in-place production of scan inputs)
    {
        int blocks = (int)(M * H_);   // 131072
        mix_kernel<<<blocks, 64, 0, stream>>>(vfirst, w0, a0, v0, k_k, k_a,
                                              k_f, v_f, wl_f, al_f, vl_f);
    }

    // ---- 5. sequential delta-rule scan, one block per (b,h)
    rwkv_scan_kernel<<<B_ * H_, 64, 0, stream>>>(r_f, wl_f, k_f, v_f, al_f, vl_f, y_f);

    // ---- 6. GroupNorm + bonus + gate -> bf16
    {
        int blocks = (int)(M * H_);
        post_kernel<<<blocks, 64, 0, stream>>>(y_f, r_f, k_f, v_f, g_f,
                                               r_k, ln_g, ln_b, zB);
    }

    // ---- 7. output projection -> d_out[0 .. B*T*C)
    gemm_wmma_bf16_kernel<<<ggrid(C_), 256, 0, stream>>>(zB, WoB, outp, (int)M, C_, C_, 0, 0);

    // ---- 8. tuple output #2: v_first pass-through
    {
        int g = (int)((MC + 255) / 256);
        copy_f32_kernel<<<g, 256, 0, stream>>>(vfirst, outp + MC, MC);
    }
}
